// SparseAutoencoder_19155554140621
// MI455X (gfx1250) — matrix operand order repaired and checked by reading, not compiled, not run on hardware
//
#include <hip/hip_runtime.h>
#include <hip/hip_bf16.h>

// ---------------------------------------------------------------------------
// TopK Sparse Autoencoder forward for MI455X (gfx1250, wave32, WMMA).
//   encode GEMM (275 GFLOP, compute bound)  -> v_wmma_f32_16x16x32_bf16,
//       128x128 block tile staged in LDS, double-buffered async-to-LDS fills
//   topk (exact radix select per row)       -> LDS hist + bitmask prefix scans
//   decode (sparse gather of W_dec columns) -> pre-transposed W_dec, L2 resident
// ---------------------------------------------------------------------------

typedef __bf16 bf16_t;
typedef bf16_t v16bf __attribute__((ext_vector_type(16)));
typedef bf16_t v8bf  __attribute__((ext_vector_type(8)));
typedef bf16_t v4bf  __attribute__((ext_vector_type(4)));
typedef float  v8f   __attribute__((ext_vector_type(8)));
typedef float  v4f   __attribute__((ext_vector_type(4)));
typedef int    v4i   __attribute__((ext_vector_type(4)));

#define N_ROWS  16384
#define D_MODEL 1024
#define D_SAE   8192
#define TOPK    32

#ifndef __has_builtin
#define __has_builtin(x) 0
#endif
#if __has_builtin(__builtin_amdgcn_global_load_async_to_lds_b128) && \
    __has_builtin(__builtin_amdgcn_s_wait_asynccnt)
#define SAE_HAS_ASYNC 1
#else
#define SAE_HAS_ASYNC 0
#endif

// 16-byte global -> LDS copy. Async (ASYNCcnt-tracked) when available.
// Builtin signature (from compiler diagnostic): (int4 AS1*, int4 AS3*, imm, imm)
__device__ __forceinline__ void cp16_g2l(const bf16_t* g, bf16_t* l) {
#if SAE_HAS_ASYNC
  void* gv = (void*)g;   // drop const, keep generic
  void* lv = (void*)l;
  __builtin_amdgcn_global_load_async_to_lds_b128(
      (__attribute__((address_space(1))) v4i*)gv,
      (__attribute__((address_space(3))) v4i*)lv, 0, 0);
#else
  *(v8bf*)l = *(const v8bf*)g;
#endif
}
#if SAE_HAS_ASYNC
#define SAE_WAIT_ASYNC(n) __builtin_amdgcn_s_wait_asynccnt(n)
#else
#define SAE_WAIT_ASYNC(n)
#endif

__device__ __forceinline__ unsigned f2k(float f) {
  unsigned u = __float_as_uint(f);
  return (u & 0x80000000u) ? ~u : (u | 0x80000000u);   // monotonic key
}
__device__ __forceinline__ float k2f(unsigned k) {
  unsigned u = (k & 0x80000000u) ? (k ^ 0x80000000u) : ~k;
  return __uint_as_float(u);
}
__device__ __forceinline__ unsigned umin_(unsigned a, unsigned b) { return a < b ? a : b; }

// ---------------------------------------------------------------------------
// Prep: W_enc f32 -> bf16
__global__ void sae_cvt_wenc(const float* __restrict__ we, bf16_t* __restrict__ web) {
  long base = 4L * (blockIdx.x * 256 + threadIdx.x);
  v4f v = *(const v4f*)(we + base);
  v4bf o;
#pragma unroll
  for (int e = 0; e < 4; ++e) o[e] = (bf16_t)v[e];
  *(v4bf*)(web + base) = o;
}

// Prep: xc = (x - b_dec) in bf16
__global__ void sae_make_xc(const float* __restrict__ x, const float* __restrict__ bdec,
                            bf16_t* __restrict__ xc) {
  long base = 4L * (blockIdx.x * 256 + threadIdx.x);
  int  k    = (int)(base & (D_MODEL - 1));
  v4f xv = *(const v4f*)(x + base);
  v4f bv = *(const v4f*)(bdec + k);
  v4bf o;
#pragma unroll
  for (int e = 0; e < 4; ++e) o[e] = (bf16_t)(xv[e] - bv[e]);
  *(v4bf*)(xc + base) = o;
}

// Prep: W_dec [1024, 8192] -> W_decT [8192, 1024]
__global__ void sae_transpose_wdec(const float* __restrict__ wd, float* __restrict__ wdT) {
  __shared__ float tile[32][33];
  int f0 = blockIdx.x * 32;
  int m0 = blockIdx.y * 32;
  int tx = threadIdx.x, ty = threadIdx.y;   // (32, 8)
#pragma unroll
  for (int i = 0; i < 32; i += 8)
    tile[ty + i][tx] = wd[(long)(m0 + ty + i) * D_SAE + f0 + tx];
  __syncthreads();
#pragma unroll
  for (int i = 0; i < 32; i += 8)
    wdT[(long)(f0 + ty + i) * D_MODEL + m0 + tx] = tile[tx][ty + i];
}

// ---------------------------------------------------------------------------
// Encoder GEMM: pre_acts = xc @ W_enc^T + b_enc  (bf16 WMMA, f32 acc)
// 128 threads = 4 waves, block tile 128x128, wave tile 64x64 (4x4 WMMA tiles).
// K staged in LDS in 32-wide chunks, double buffered, async fills.
__global__ void __launch_bounds__(128)
sae_enc_gemm(const bf16_t* __restrict__ xc, const bf16_t* __restrict__ web,
             const float* __restrict__ benc, float* __restrict__ pre) {
  constexpr int LROW  = 40;           // 32 k-elems padded to 40 (80B, 16B-aligned rows)
  constexpr int ASZ   = 128 * LROW;   // one A (or B) tile in bf16 elems
  constexpr int BUFSZ = 2 * ASZ;      // A + B per buffer
  __shared__ bf16_t lds[2 * BUFSZ];   // 40 KB double-buffered

  const int tid   = threadIdx.x;
  const int lane  = tid & 31;
  const int wave  = tid >> 5;
  const int waveM = wave & 1, waveN = wave >> 1;
  const int hl    = lane >> 4;        // lane half
  const int mr    = lane & 15;
  const long rowBlock = (long)blockIdx.x * 128;
  const long colBlock = (long)blockIdx.y * 128;
  const bf16_t* gA = xc + rowBlock * D_MODEL;
  const bf16_t* gB = web + colBlock * D_MODEL;

  v8f acc[4][4];
#pragma unroll
  for (int t = 0; t < 4; ++t)
#pragma unroll
    for (int u = 0; u < 4; ++u)
#pragma unroll
      for (int r = 0; r < 8; ++r) acc[t][u][r] = 0.0f;

  // Cooperative fill of one 128x32 A tile + 128x32 B tile (8 x b128 per thread).
  auto fill = [&](int bb, int k0) {
#pragma unroll
    for (int i = 0; i < 4; ++i) {
      const int chunk = i * 128 + tid;       // 512 chunks of 16B per tile
      const int row = chunk >> 2, c = chunk & 3;
      cp16_g2l(gA + (long)row * D_MODEL + k0 + c * 8,
               &lds[bb * BUFSZ + row * LROW + c * 8]);
      cp16_g2l(gB + (long)row * D_MODEL + k0 + c * 8,
               &lds[bb * BUFSZ + ASZ + row * LROW + c * 8]);
    }
  };

  fill(0, 0);
  constexpr int NK = D_MODEL / 32;
  for (int ks = 0; ks < NK; ++ks) {
    const int cur = ks & 1;
    if (ks + 1 < NK) {
      fill(cur ^ 1, (ks + 1) * 32);   // prefetch next buffer
      SAE_WAIT_ASYNC(8);              // current buffer's 8 copies complete
    } else {
      SAE_WAIT_ASYNC(0);
    }
    __syncthreads();

    // A 16x32 layout: lane half h holds K = h*8 + {0..7} and h*8 + {16..23}
    const bf16_t* Ab = &lds[cur * BUFSZ + (waveM * 64 + mr) * LROW + hl * 8];
    // B 32x16 layout: lane half h holds K = h*8 + {0..7} and h*8 + {16..23}
    const bf16_t* Bb = &lds[cur * BUFSZ + ASZ + (waveN * 64 + mr) * LROW + hl * 8];
    union Frag { v16bf v; v8bf h[2]; } A[4], B[4];
#pragma unroll
    for (int t = 0; t < 4; ++t) {
      A[t].h[0] = *(const v8bf*)(Ab + t * 16 * LROW);
      A[t].h[1] = *(const v8bf*)(Ab + t * 16 * LROW + 16);
    }
#pragma unroll
    for (int u = 0; u < 4; ++u) {
      B[u].h[0] = *(const v8bf*)(Bb + u * 16 * LROW);
      B[u].h[1] = *(const v8bf*)(Bb + u * 16 * LROW + 16);
    }
#pragma unroll
    for (int t = 0; t < 4; ++t)
#pragma unroll
      for (int u = 0; u < 4; ++u)
        acc[t][u] = __builtin_amdgcn_wmma_f32_16x16x32_bf16(
            false, A[t].v, false, B[u].v, (short)0, acc[t][u], false, false);
    __syncthreads();   // all waves done reading `cur` before it is refilled
  }

  // D layout: lanes 0-15 VGPR r -> M=r; lanes 16-31 VGPR r -> M=8+r; N = mr
#pragma unroll
  for (int u = 0; u < 4; ++u) {
    const long col  = colBlock + waveN * 64 + u * 16 + mr;
    const float bias = benc[col];
#pragma unroll
    for (int t = 0; t < 4; ++t) {
      const long row0 = rowBlock + waveM * 64 + t * 16 + hl * 8;
#pragma unroll
      for (int r = 0; r < 8; ++r)
        pre[(row0 + r) * D_SAE + col] = acc[t][u][r] + bias;
    }
  }
}

// ---------------------------------------------------------------------------
// Exact TopK per row (K=32 of 8192). One block per row; thread t owns elements
// [t*32, t*32+32). 4-pass MSB radix select, deterministic tie-break by lowest
// index. Rewrites features in place; emits (idx,val) pairs + positive count.
__global__ void __launch_bounds__(256)
sae_topk(float* __restrict__ feats, int* __restrict__ tIdx, float* __restrict__ tVal,
         unsigned* __restrict__ l0P) {
  const int row = blockIdx.x;
  const int tid = threadIdx.x;
  const long rowOff = (long)row * D_SAE + tid * 32;

  unsigned keys[TOPK];
#pragma unroll
  for (int q = 0; q < 8; ++q) {
    v4f v = *(const v4f*)(feats + rowOff + q * 4);
#pragma unroll
    for (int e = 0; e < 4; ++e) keys[q * 4 + e] = f2k(v[e]);
  }

  __shared__ unsigned hist[256];
  __shared__ unsigned sh_bin, sh_rem;

  unsigned prefix = 0, remaining = TOPK;
  for (int pass = 3; pass >= 0; --pass) {
    const int shift = pass * 8;
    const unsigned pmask = (pass == 3) ? 0u : ~((1u << (8 * (pass + 1))) - 1u);
    hist[tid] = 0;
    __syncthreads();
#pragma unroll
    for (int j = 0; j < TOPK; ++j) {
      unsigned k = keys[j];
      if ((k & pmask) == prefix) atomicAdd(&hist[(k >> shift) & 0xFFu], 1u);
    }
    __syncthreads();
    if (tid == 0) {
      unsigned cumBefore = 0;
      int b = 255;
      for (;;) {
        unsigned c = hist[b];
        if (cumBefore + c >= remaining || b == 0) break;
        cumBefore += c;
        --b;
      }
      sh_bin = (unsigned)b;
      sh_rem = remaining - cumBefore;
    }
    __syncthreads();
    prefix |= (sh_bin << shift);
    remaining = sh_rem;
  }
  const unsigned T = prefix;
  const unsigned need = remaining;

  unsigned gw = 0, tw = 0;
#pragma unroll
  for (int j = 0; j < TOPK; ++j) {
    unsigned k = keys[j];
    gw |= (unsigned)(k > T) << j;
    tw |= (unsigned)(k == T) << j;
  }

  __shared__ unsigned gpre[256], tpre[256];
  gpre[tid] = (unsigned)__popc(gw);
  tpre[tid] = (unsigned)__popc(tw);
  __syncthreads();
  for (int off = 1; off < 256; off <<= 1) {
    unsigned ga = (tid >= off) ? gpre[tid - off] : 0u;
    unsigned ta = (tid >= off) ? tpre[tid - off] : 0u;
    __syncthreads();
    gpre[tid] += ga;
    tpre[tid] += ta;
    __syncthreads();
  }
  const unsigned gExcl = gpre[tid] - (unsigned)__popc(gw);
  const unsigned tExcl = tpre[tid] - (unsigned)__popc(tw);

  unsigned allowed = (need > tExcl) ? umin_(need - tExcl, (unsigned)__popc(tw)) : 0u;
  unsigned kw = gw, tmp = tw;
  for (unsigned i = 0; i < allowed; ++i) { kw |= tmp & (0u - tmp); tmp &= tmp - 1u; }
  const unsigned base = gExcl + umin_(tExcl, need);

  unsigned posCnt = 0;
#pragma unroll
  for (int q = 0; q < 8; ++q) {
    v4f o;
#pragma unroll
    for (int e = 0; e < 4; ++e) {
      const int j = q * 4 + e;
      const float v = k2f(keys[j]);
      const bool kept = (kw >> j) & 1u;
      const float relu = v > 0.0f ? v : 0.0f;
      o[e] = kept ? relu : 0.0f;
      if (kept) {
        unsigned pos = base + (unsigned)__popc(kw & ((1u << j) - 1u));
        tIdx[(long)row * TOPK + pos] = tid * 32 + j;
        tVal[(long)row * TOPK + pos] = relu;
        if (v > 0.0f) posCnt++;
      }
    }
    *(v4f*)(feats + rowOff + q * 4) = o;
  }

  __shared__ unsigned redc[256];
  redc[tid] = posCnt;
  __syncthreads();
  for (int off = 128; off > 0; off >>= 1) {
    if (tid < off) redc[tid] += redc[tid + off];
    __syncthreads();
  }
  if (tid == 0) l0P[row] = redc[0];
}

// ---------------------------------------------------------------------------
// Sparse decode: x_hat[row,:] = b_dec + sum_j val_j * W_decT[idx_j,:]
__global__ void __launch_bounds__(256)
sae_decode(const float* __restrict__ x, const float* __restrict__ bdec,
           const float* __restrict__ wdT, const int* __restrict__ tIdx,
           const float* __restrict__ tVal, float* __restrict__ xhat,
           float* __restrict__ sseP) {
  const int row = blockIdx.x;
  const int tid = threadIdx.x;
  const int m = tid * 4;
  v4f acc = *(const v4f*)(bdec + m);
#pragma unroll 4
  for (int j = 0; j < TOPK; ++j) {
    const int   f = tIdx[(long)row * TOPK + j];
    const float v = tVal[(long)row * TOPK + j];
    if (v != 0.0f) {
      v4f w = *(const v4f*)(wdT + (long)f * D_MODEL + m);
      acc += v * w;
    }
  }
  *(v4f*)(xhat + (long)row * D_MODEL + m) = acc;
  v4f xv = *(const v4f*)(x + (long)row * D_MODEL + m);
  v4f d = acc - xv;
  float s = d[0] * d[0] + d[1] * d[1] + d[2] * d[2] + d[3] * d[3];

  __shared__ float red[256];
  red[tid] = s;
  __syncthreads();
  for (int off = 128; off > 0; off >>= 1) {
    if (tid < off) red[tid] += red[tid + off];
    __syncthreads();
  }
  if (tid == 0) sseP[row] = red[0];
}

// ---------------------------------------------------------------------------
// Deterministic fixed-order final reduction of loss / l0 scalars.
__global__ void __launch_bounds__(256)
sae_finalize(const float* __restrict__ sseP, const unsigned* __restrict__ l0P,
             float* __restrict__ scal) {
  const int tid = threadIdx.x;
  float s = 0.0f;
  unsigned c = 0;
  for (int i = tid; i < N_ROWS; i += 256) { s += sseP[i]; c += l0P[i]; }
  __shared__ float rs[256];
  __shared__ unsigned rc[256];
  rs[tid] = s; rc[tid] = c;
  __syncthreads();
  for (int off = 128; off > 0; off >>= 1) {
    if (tid < off) { rs[tid] += rs[tid + off]; rc[tid] += rc[tid + off]; }
    __syncthreads();
  }
  if (tid == 0) {
    const float mse = rs[0] / ((float)N_ROWS * (float)D_MODEL);
    scal[0] = mse;                           // loss (TopK mode: loss == mse)
    scal[1] = mse;                           // mse_loss
    scal[2] = 0.0f;                          // l1_loss
    scal[3] = (float)rc[0] / (float)N_ROWS;  // l0
  }
}

// ---------------------------------------------------------------------------
extern "C" void kernel_launch(void* const* d_in, const int* in_sizes, int n_in,
                              void* d_out, int out_size, void* d_ws, size_t ws_size,
                              hipStream_t stream) {
  (void)in_sizes; (void)n_in; (void)out_size; (void)ws_size;
  const float* x     = (const float*)d_in[0];
  const float* W_enc = (const float*)d_in[1];
  const float* b_enc = (const float*)d_in[2];
  const float* W_dec = (const float*)d_in[3];
  const float* b_dec = (const float*)d_in[4];

  float* xhat  = (float*)d_out;
  float* feats = xhat + (long)N_ROWS * D_MODEL;
  float* scal  = feats + (long)N_ROWS * D_SAE;

  char* w = (char*)d_ws;
  bf16_t*   xc   = (bf16_t*)w;   w += (size_t)N_ROWS * D_MODEL * 2;
  bf16_t*   web  = (bf16_t*)w;   w += (size_t)D_SAE * D_MODEL * 2;
  float*    wdT  = (float*)w;    w += (size_t)D_SAE * D_MODEL * 4;
  int*      tIdx = (int*)w;      w += (size_t)N_ROWS * TOPK * 4;
  float*    tVal = (float*)w;    w += (size_t)N_ROWS * TOPK * 4;
  float*    sseP = (float*)w;    w += (size_t)N_ROWS * 4;
  unsigned* l0P  = (unsigned*)w; w += (size_t)N_ROWS * 4;

  sae_cvt_wenc<<<(D_SAE * D_MODEL / 4) / 256, 256, 0, stream>>>(W_enc, web);
  sae_make_xc<<<(N_ROWS * D_MODEL / 4) / 256, 256, 0, stream>>>(x, b_dec, xc);
  sae_transpose_wdec<<<dim3(D_SAE / 32, D_MODEL / 32), dim3(32, 8), 0, stream>>>(W_dec, wdT);
  sae_enc_gemm<<<dim3(N_ROWS / 128, D_SAE / 128), 128, 0, stream>>>(xc, web, b_enc, feats);
  sae_topk<<<N_ROWS, 256, 0, stream>>>(feats, tIdx, tVal, l0P);
  sae_decode<<<N_ROWS, 256, 0, stream>>>(x, b_dec, wdT, tIdx, tVal, xhat, sseP);
  sae_finalize<<<1, 256, 0, stream>>>(sseP, l0P, scal);
}
